// SpatialEdgeEncoder_45672682225971
// MI455X (gfx1250) — matrix operand order repaired and checked by reading, not compiled, not run on hardware
//
#include <hip/hip_runtime.h>
#include <hip/hip_bf16.h>

typedef __attribute__((ext_vector_type(8)))  __bf16 v8bf;
typedef __attribute__((ext_vector_type(16))) __bf16 v16bf;
typedef __attribute__((ext_vector_type(8)))  float  v8f;

#define DIMD 256
#define KN   64
#define AB_STRIDE 264   // padded LDS row stride in bf16 elems (528 B = 132 dwords)

__device__ __forceinline__ unsigned short f2bf(float x) {
    unsigned int u = __float_as_uint(x);
    unsigned int r = u + 0x7FFFu + ((u >> 16) & 1u);   // round-to-nearest-even
    return (unsigned short)(r >> 16);
}

__device__ __forceinline__ float gelu_exact(float x) {
    return 0.5f * x * (1.0f + erff(x * 0.70710678118654752f));
}

// ---------------------------------------------------------------------------
// Prep 1: W2 (f32 row-major [K=256][N=256]) -> bf16 row-major in workspace.
// Stored transposed (bf16 [N][K]) so a B-fragment lane (column n) reads
// contiguous K values, same chunk layout as A.
// ---------------------------------------------------------------------------
__global__ void prep_w2_kernel(const float* __restrict__ W2,
                               unsigned short* __restrict__ w2bf) {
    int idx = blockIdx.x * blockDim.x + threadIdx.x;   // 0..65535
    w2bf[(idx & 255) * DIMD + (idx >> 8)] = f2bf(W2[idx]);
}

// ---------------------------------------------------------------------------
// Prep 2: P[b,k,d] = sum_c feat[b,k,c] * W1[(4+c), d]   (f32, tiny GEMM)
// One block per (b,k); thread t owns output column d = t.
// ---------------------------------------------------------------------------
__global__ void prep_p_kernel(const float* __restrict__ feat,
                              const float* __restrict__ W1,
                              float* __restrict__ P) {
    __shared__ float f[DIMD];
    const int bi = blockIdx.x;
    const int t  = threadIdx.x;
    f[t] = feat[bi * DIMD + t];
    __syncthreads();
    const float* w = W1 + 4 * DIMD + t;   // W1[(4+c)*256 + t]
    float s = 0.0f;
#pragma unroll 8
    for (int c = 0; c < DIMD; ++c) s = fmaf(f[c], w[c * DIMD], s);
    P[bi * DIMD + t] = s;
}

// ---------------------------------------------------------------------------
// Main kernel: one workgroup per (b,i); 256 threads = 8 wave32.
//   h[j,d]  = geo(i,j).W1geo[:,d] + P[b,i,d] - P[b,j,d] + b1[d]
//   out     = GELU(h) @ W2 + b2   via v_wmma_f32_16x16x32_bf16
// ---------------------------------------------------------------------------
__global__ void __launch_bounds__(256)
edge_main_kernel(const float* __restrict__ pos,
                 const float* __restrict__ W1,
                 const float* __restrict__ b1,
                 const float* __restrict__ b2,
                 const unsigned short* __restrict__ w2bf,
                 const float* __restrict__ P,
                 float* __restrict__ out) {
    __shared__ float geo[KN][4];
    __shared__ __attribute__((aligned(16))) unsigned short Abuf[KN * AB_STRIDE];

    const int wg = blockIdx.x;       // b*64 + i
    const int b  = wg >> 6;
    const int i  = wg & 63;
    const int t  = threadIdx.x;

    // ---- Phase 0: geometric features for all j (computed once) -------------
    if (t < KN) {
        const float xi = pos[(b * KN + i) * 2 + 0];
        const float yi = pos[(b * KN + i) * 2 + 1];
        const float xj = pos[(b * KN + t) * 2 + 0];
        const float yj = pos[(b * KN + t) * 2 + 1];
        const float dx = xi - xj, dy = yi - yj;
        geo[t][0] = dx;
        geo[t][1] = dy;
        geo[t][2] = sqrtf(dx * dx + dy * dy);
        geo[t][3] = atan2f(dy, dx);
    }
    __syncthreads();

    // ---- Phase 1: h -> GELU -> bf16 A-tile in LDS (thread owns column d=t) -
    {
        const int   d   = t;
        const float w0  = W1[0 * DIMD + d];
        const float w1  = W1[1 * DIMD + d];
        const float w2v = W1[2 * DIMD + d];
        const float w3  = W1[3 * DIMD + d];
        const float bse = P[wg * DIMD + d] + b1[d];   // P[b,i,d] + b1[d]
        const float* Pj = P + (size_t)(b * KN) * DIMD + d;
#pragma unroll 4
        for (int j = 0; j < KN; ++j) {
            float x = bse - Pj[(size_t)j * DIMD]
                    + geo[j][0] * w0 + geo[j][1] * w1
                    + geo[j][2] * w2v + geo[j][3] * w3;
            Abuf[j * AB_STRIDE + d] = f2bf(gelu_exact(x));
        }
    }
    __syncthreads();

    // ---- Phase 2: out[64 x 256] = A(bf16) @ W2(bf16) + b2, f32 accumulate --
    const int wave = t >> 5;
    const int lane = t & 31;
    const int l15  = lane & 15;
    const int hi   = lane >> 4;
    const int n0   = wave * 32;          // 32-column slab per wave

    v8f acc[4][2];
#pragma unroll
    for (int m = 0; m < 4; ++m)
#pragma unroll
        for (int n = 0; n < 2; ++n)
            acc[m][n] = (v8f){0.f, 0.f, 0.f, 0.f, 0.f, 0.f, 0.f, 0.f};

#pragma unroll 2
    for (int ks = 0; ks < 8; ++ks) {
        const int k0 = ks * 32;

        // A fragments: 16-bit A 16x32 layout — lane m holds row m; lo half
        // K = hi*8 .. hi*8+7, hi half K = 16+hi*8 .. (two 16B chunks, +32B apart)
        v16bf afrag[4];
#pragma unroll
        for (int m = 0; m < 4; ++m) {
            const unsigned short* ap =
                Abuf + (m * 16 + l15) * AB_STRIDE + k0 + hi * 8;
            v8bf lo  = *(const v8bf*)ap;
            v8bf hi8 = *(const v8bf*)(ap + 16);
            afrag[m] = __builtin_shufflevector(lo, hi8,
                0, 1, 2, 3, 4, 5, 6, 7, 8, 9, 10, 11, 12, 13, 14, 15);
        }

        // B fragments: lane n holds column n; K = hi*8.., 16+hi*8.. (two 16B chunks)
        v16bf bfrag[2];
#pragma unroll
        for (int n = 0; n < 2; ++n) {
            const unsigned short* bp =
                w2bf + (size_t)(n0 + n * 16 + l15) * DIMD + k0 + hi * 8;
            v8bf lo  = *(const v8bf*)bp;
            v8bf hi8 = *(const v8bf*)(bp + 16);
            bfrag[n] = __builtin_shufflevector(lo, hi8,
                0, 1, 2, 3, 4, 5, 6, 7, 8, 9, 10, 11, 12, 13, 14, 15);
        }

#pragma unroll
        for (int m = 0; m < 4; ++m)
#pragma unroll
            for (int n = 0; n < 2; ++n)
                acc[m][n] = __builtin_amdgcn_wmma_f32_16x16x32_bf16(
                    false, afrag[m], false, bfrag[n],
                    (short)0, acc[m][n], false, false);
    }

    // ---- Store: C layout — VGPR r: lanes0-15 -> M=r, lanes16-31 -> M=8+r ---
    float* obase = out + (size_t)wg * (KN * DIMD);
#pragma unroll
    for (int m = 0; m < 4; ++m)
#pragma unroll
        for (int n = 0; n < 2; ++n) {
            const int   d    = n0 + n * 16 + l15;
            const float bias = b2[d];
#pragma unroll
            for (int r = 0; r < 8; ++r) {
                const int row = m * 16 + hi * 8 + r;
                obase[(size_t)row * DIMD + d] = acc[m][n][r] + bias;
            }
        }
}

// ---------------------------------------------------------------------------
// Launch.  ws layout: [0,128KB): bf16 W2 ; [128KB, 128KB+2MB): f32 P
// ---------------------------------------------------------------------------
extern "C" void kernel_launch(void* const* d_in, const int* in_sizes, int n_in,
                              void* d_out, int out_size, void* d_ws, size_t ws_size,
                              hipStream_t stream) {
    const float* feat = (const float*)d_in[0];   // [32,64,256]
    const float* pos  = (const float*)d_in[1];   // [32,64,2]
    const float* W1   = (const float*)d_in[2];   // [260,256]
    const float* b1   = (const float*)d_in[3];   // [256]
    const float* W2   = (const float*)d_in[4];   // [256,256]
    const float* b2   = (const float*)d_in[5];   // [256]
    float* out = (float*)d_out;                  // [32,64,64,256]

    unsigned short* w2bf = (unsigned short*)d_ws;
    float* P = (float*)((char*)d_ws + (size_t)DIMD * DIMD * sizeof(unsigned short));

    prep_w2_kernel<<<DIMD, DIMD, 0, stream>>>(W2, w2bf);
    prep_p_kernel<<<32 * KN, DIMD, 0, stream>>>(feat, W1, P);
    edge_main_kernel<<<32 * KN, 256, 0, stream>>>(pos, W1, b1, b2, w2bf, P, out);
}
